// Mamba3Block_23957327577557
// MI455X (gfx1250) — compile-verified, matrix operand order checked by reading, not run on hardware
//
#include <hip/hip_runtime.h>
#include <hip/hip_bf16.h>

// ---------------- constants (match reference) ----------------
constexpr int DIM      = 2048;
constexpr int D_STATE  = 128;
constexpr int HEADDIM  = 64;
constexpr int CHUNK    = 64;
constexpr int D_INNER  = 4096;
constexpr int NHEADS   = 64;
constexpr int D_PROJ   = 2 * D_INNER + 2 * D_STATE + NHEADS;  // 8512
constexpr int BATCH    = 2;
constexpr int SEQ      = 4096;
constexpr int NCHUNK   = SEQ / CHUNK;                          // 64
constexpr int NTOK     = BATCH * SEQ;                          // 8192

typedef __attribute__((ext_vector_type(16))) _Float16 v16h;
typedef __attribute__((ext_vector_type(8)))  float    v8f;
typedef __attribute__((ext_vector_type(4)))  int      v4i;

#define AS1 __attribute__((address_space(1)))
#define AS3 __attribute__((address_space(3)))

// ---------------- CDNA5 async global->LDS copy (ASYNCcnt path) --------------
// Builtin signature (from hipcc diagnostic): param 1 is `int4 __device__*`
// (global AS, non-const).  Param 2 is the LDS-side pointer.
__device__ __forceinline__ void async_copy16(void* lds_dst, const void* gsrc) {
#if __has_builtin(__builtin_amdgcn_global_load_async_to_lds_b128)
  __builtin_amdgcn_global_load_async_to_lds_b128(
      (AS1 v4i*)(unsigned long long)gsrc,
      (AS3 v4i*)(unsigned)(unsigned long long)lds_dst, 0, 0);
#else
  unsigned lo = (unsigned)(unsigned long long)lds_dst;
  asm volatile("global_load_async_to_lds_b128 %0, %1, off"
               :: "v"(lo), "v"(gsrc) : "memory");
#endif
}

__device__ __forceinline__ void async_wait0() {
#if __has_builtin(__builtin_amdgcn_s_wait_asynccnt)
  __builtin_amdgcn_s_wait_asynccnt(0);
#else
  asm volatile("s_wait_asynccnt 0" ::: "memory");
#endif
}

// ---------------- WMMA helpers (CDNA5 gfx1250, wave32) ----------------
__device__ __forceinline__ v8f wmma16(v16h a, v16h b, v8f c) {
  // D = A(16x32 f16) * B(32x16 f16) + C(16x16 f32)
  return __builtin_amdgcn_wmma_f32_16x16x32_f16(false, a, false, b, (short)0, c,
                                                false, false);
}

// Load one 16x32 fragment.  For the A-matrix: (lane&15) selects row M,
// K runs over {kb..kb+7, kb+16..kb+23} with kb = 8*(lane>=16).
// For the B-matrix the same layout holds with (lane&15) selecting column N.
// base points at element [sel=0][k=0]; memory offset = sel*sel_stride + k*k_stride.
__device__ __forceinline__ v16h load_frag(const _Float16* base, int sel_stride,
                                          int k_stride) {
  const int lane = threadIdx.x & 31;
  const int kb = (lane >> 4) << 3;  // 0 or 8
  const _Float16* p = base + (lane & 15) * sel_stride;
  v16h f;
#pragma unroll
  for (int i = 0; i < 8; ++i) {
    f[i]     = p[(kb + i) * k_stride];
    f[8 + i] = p[(kb + 16 + i) * k_stride];
  }
  return f;
}

// ---------------- kernel 1: layernorm(u) -> f16 ----------------
__global__ __launch_bounds__(256) void layernorm_kernel(
    const float* __restrict__ u, const float* __restrict__ w,
    const float* __restrict__ bq, _Float16* __restrict__ out) {
  __shared__ float rs[256], rq[256];
  const int t = blockIdx.x;
  const float* row = u + (size_t)t * DIM;
  float s = 0.f, q = 0.f;
  for (int e = threadIdx.x; e < DIM; e += 256) {
    float v = row[e];
    s += v;
    q += v * v;
  }
  rs[threadIdx.x] = s;
  rq[threadIdx.x] = q;
  __syncthreads();
  for (int o = 128; o > 0; o >>= 1) {
    if (threadIdx.x < o) {
      rs[threadIdx.x] += rs[threadIdx.x + o];
      rq[threadIdx.x] += rq[threadIdx.x + o];
    }
    __syncthreads();
  }
  const float mu = rs[0] / (float)DIM;
  const float var = rq[0] / (float)DIM - mu * mu;
  const float inv = rsqrtf(var + 1e-5f);
  for (int e = threadIdx.x; e < DIM; e += 256)
    out[(size_t)t * DIM + e] = (_Float16)((row[e] - mu) * inv * w[e] + bq[e]);
}

// ---------------- kernel 2: f32 -> f16 convert ----------------
__global__ __launch_bounds__(256) void cvt_f16_kernel(const float* __restrict__ in,
                                                      _Float16* __restrict__ o,
                                                      int n) {
  int i = blockIdx.x * 256 + threadIdx.x;
  if (i < n) o[i] = (_Float16)in[i];
}

// ---------------- kernel 3/8: WMMA GEMM  C[M,N] = A[M,K] * Bw[N,K]^T (+resid)
__global__ __launch_bounds__(128) void gemm_f16_kernel(
    const _Float16* __restrict__ A, const _Float16* __restrict__ Bw,
    const float* __restrict__ resid, float* __restrict__ C, int M, int N, int K) {
  const int wv = threadIdx.x >> 5;
  const int m0 = blockIdx.y * 64 + (wv >> 1) * 32;
  const int n0 = blockIdx.x * 64 + (wv & 1) * 32;
  v8f acc[2][2] = {};
  for (int kt = 0; kt < K; kt += 32) {
    v16h a0 = load_frag(A + (size_t)m0 * K + kt, K, 1);
    v16h a1 = load_frag(A + (size_t)(m0 + 16) * K + kt, K, 1);
    v16h b0 = load_frag(Bw + (size_t)n0 * K + kt, K, 1);
    v16h b1 = load_frag(Bw + (size_t)(n0 + 16) * K + kt, K, 1);
    acc[0][0] = wmma16(a0, b0, acc[0][0]);
    acc[0][1] = wmma16(a0, b1, acc[0][1]);
    acc[1][0] = wmma16(a1, b0, acc[1][0]);
    acc[1][1] = wmma16(a1, b1, acc[1][1]);
  }
  const int lane = threadIdx.x & 31;
  const int rofs = (lane >> 4) * 8, cofs = lane & 15;
#pragma unroll
  for (int i = 0; i < 2; ++i)
#pragma unroll
    for (int j = 0; j < 2; ++j)
#pragma unroll
      for (int r = 0; r < 8; ++r) {
        int row = m0 + i * 16 + rofs + r;
        int col = n0 + j * 16 + cofs;
        float v = acc[i][j][r];
        if (resid) v += resid[(size_t)row * N + col];
        C[(size_t)row * N + col] = v;
      }
}

// ---------------- kernel 4: split proj; rmsnorm B,C; softplus dt -> dA ------
__global__ __launch_bounds__(128) void extract_kernel(
    const float* __restrict__ proj, const float* __restrict__ bnw,
    const float* __restrict__ cnw, const float* __restrict__ dtb,
    const float* __restrict__ Alog, _Float16* __restrict__ xh,
    _Float16* __restrict__ Bh, _Float16* __restrict__ Ch,
    float* __restrict__ dAo) {
  const int t = blockIdx.x, tid = threadIdx.x;
  const float* p = proj + (size_t)t * D_PROJ;
  for (int e = tid; e < D_INNER; e += 128)
    xh[(size_t)t * D_INNER + e] = (_Float16)p[D_INNER + e];
  __shared__ float red[128];
  // B rmsnorm (128 state dims, one per thread)
  float bv = p[2 * D_INNER + tid];
  red[tid] = bv * bv;
  __syncthreads();
  for (int o = 64; o > 0; o >>= 1) {
    if (tid < o) red[tid] += red[tid + o];
    __syncthreads();
  }
  float rb = rsqrtf(red[0] / (float)D_STATE + 1e-5f);
  __syncthreads();
  Bh[(size_t)t * D_STATE + tid] = (_Float16)(bv * rb * bnw[tid]);
  // C rmsnorm
  float cv = p[2 * D_INNER + D_STATE + tid];
  red[tid] = cv * cv;
  __syncthreads();
  for (int o = 64; o > 0; o >>= 1) {
    if (tid < o) red[tid] += red[tid + o];
    __syncthreads();
  }
  float rc = rsqrtf(red[0] / (float)D_STATE + 1e-5f);
  Ch[(size_t)t * D_STATE + tid] = (_Float16)(cv * rc * cnw[tid]);
  if (tid < NHEADS) {
    float d = p[2 * D_INNER + 2 * D_STATE + tid] + dtb[tid];
    float sp = d > 20.f ? d : log1pf(expf(d));
    dAo[(size_t)t * NHEADS + tid] = sp * (-expf(Alog[tid]));
  }
}

// ---------------- kernel 5: per-chunk SSD (Y_diag + states) -----------------
// block = (b, c, h); 128 threads = 4 waves.  x and B tiles are staged into
// LDS with CDNA5 async global->LDS b128 copies (ASYNCcnt), then all strided
// fragment reads come from LDS.  decay_states is folded into the x A-fragments.
__global__ __launch_bounds__(128) void chunk_kernel(
    const _Float16* __restrict__ xh, const _Float16* __restrict__ Bhp,
    const _Float16* __restrict__ Chp, const float* __restrict__ dA,
    float* __restrict__ AcsOut, float* __restrict__ ydiag,
    float* __restrict__ states) {
  const int h = blockIdx.x, c = blockIdx.y, b = blockIdx.z;
  const int tid = threadIdx.x, lane = tid & 31, wv = tid >> 5;
  const int t0 = b * SEQ + c * CHUNK;
  __shared__ float sAcs[CHUNK];
  __shared__ float sDec[CHUNK];
  __shared__ _Float16 sX[CHUNK * HEADDIM];   // 8KB: x[j][p] head slice
  __shared__ _Float16 sB[CHUNK * D_STATE];   // 16KB: B[j][n]
  __shared__ _Float16 sM[CHUNK * CHUNK];     // 8KB: masked CB^T

  // ---- issue async tile copies (overlap with cumsum below) ----
#pragma unroll
  for (int k = 0; k < 4; ++k) {  // sX: 512 x 16B
    int idx = tid + k * 128;
    int row = idx >> 3, seg = idx & 7;
    async_copy16(sX + row * HEADDIM + seg * 8,
                 xh + (size_t)(t0 + row) * D_INNER + h * HEADDIM + seg * 8);
  }
#pragma unroll
  for (int k = 0; k < 8; ++k) {  // sB: 1024 x 16B
    int idx = tid + k * 128;
    int row = idx >> 4, seg = idx & 15;
    async_copy16(sB + row * D_STATE + seg * 8,
                 Bhp + (size_t)(t0 + row) * D_STATE + seg * 8);
  }
  if (tid < CHUNK) sAcs[tid] = dA[(size_t)(t0 + tid) * NHEADS + h];
  __syncthreads();
  if (tid == 0) {  // inclusive cumsum of dA over the chunk
    float s = 0.f;
    for (int i = 0; i < CHUNK; ++i) { s += sAcs[i]; sAcs[i] = s; }
  }
  async_wait0();   // each wave drains its async copies before the barrier
  __syncthreads(); // sX, sB, sAcs now valid for everyone
  if (tid < CHUNK) sDec[tid] = expf(sAcs[CHUNK - 1] - sAcs[tid]);

  // ---- M = C * B^T  (64x64, K = 128 states); B operand from LDS ----
  const int iT = wv * 16;
  v8f accM[4] = {};
  for (int kt = 0; kt < D_STATE; kt += 32) {
    v16h aC = load_frag(Chp + (size_t)(t0 + iT) * D_STATE + kt, D_STATE, 1);
#pragma unroll
    for (int jt = 0; jt < 4; ++jt) {
      v16h bB = load_frag(sB + jt * 16 * D_STATE + kt, D_STATE, 1);
      accM[jt] = wmma16(aC, bB, accM[jt]);
    }
  }
  const int rofs = (lane >> 4) * 8, cofs = lane & 15;
#pragma unroll
  for (int jt = 0; jt < 4; ++jt)
#pragma unroll
    for (int r = 0; r < 8; ++r) {
      int i = iT + rofs + r;
      int j = jt * 16 + cofs;
      float v = (j <= i) ? accM[jt][r] * expf(sAcs[i] - sAcs[j]) : 0.f;
      sM[i * CHUNK + j] = (_Float16)v;
    }
  __syncthreads();  // publish sM (and sDec)

  // ---- Y_diag = (M.L) * x  (64x64, K = 64); x operand from LDS ----
  v8f accY[4] = {};
  for (int kt = 0; kt < CHUNK; kt += 32) {
    v16h aM = load_frag(sM + iT * CHUNK + kt, CHUNK, 1);
#pragma unroll
    for (int pt = 0; pt < 4; ++pt) {
      v16h bx = load_frag(sX + kt * HEADDIM + pt * 16, 1, HEADDIM);
      accY[pt] = wmma16(aM, bx, accY[pt]);
    }
  }
#pragma unroll
  for (int pt = 0; pt < 4; ++pt)
#pragma unroll
    for (int r = 0; r < 8; ++r) {
      int i = iT + rofs + r, p = pt * 16 + cofs;
      ydiag[(size_t)(t0 + i) * D_INNER + h * HEADDIM + p] = accY[pt][r];
    }

  // ---- states = (x*decay)^T * B  (64p x 128n, K = 64) ----
  const int pT = wv * 16;
  const int kb = (lane >> 4) << 3;
  v8f accS[8] = {};
  for (int kt = 0; kt < CHUNK; kt += 32) {
    v16h aX = load_frag(sX + kt * HEADDIM + pT, 1, HEADDIM);
#pragma unroll
    for (int i = 0; i < 8; ++i) {  // fold decay_states[j] into A fragment
      aX[i]     = (_Float16)((float)aX[i]     * sDec[kt + kb + i]);
      aX[8 + i] = (_Float16)((float)aX[8 + i] * sDec[kt + kb + 16 + i]);
    }
#pragma unroll
    for (int nt = 0; nt < 8; ++nt) {
      v16h bS = load_frag(sB + kt * D_STATE + nt * 16, 1, D_STATE);
      accS[nt] = wmma16(aX, bS, accS[nt]);
    }
  }
  const size_t sbase = (size_t)((b * NCHUNK + c) * NHEADS + h) * (HEADDIM * D_STATE);
#pragma unroll
  for (int nt = 0; nt < 8; ++nt)
#pragma unroll
    for (int r = 0; r < 8; ++r) {
      int p = pT + rofs + r, n = nt * 16 + cofs;
      states[sbase + p * D_STATE + n] = accS[nt][r];
    }
  if (tid < CHUNK)
    AcsOut[(size_t)((b * NCHUNK + c) * NHEADS + h) * CHUNK + tid] = sAcs[tid];
}

// ---------------- kernel 6: inter-chunk state recurrence --------------------
// new_states[c] = state entering chunk c:  S[0]=0; S[c+1] = exp(Aend[c])*S[c] + states[c]
__global__ __launch_bounds__(256) void scan_kernel(const float* __restrict__ states,
                                                   const float* __restrict__ Acs,
                                                   float* __restrict__ ns) {
  const int t = blockIdx.x * 256 + threadIdx.x;  // over b*h*p*n = 2*64*64*128
  const int n = t & 127, p = (t >> 7) & 63, h = (t >> 13) & 63, b = t >> 19;
  float s = 0.f;
  for (int c = 0; c < NCHUNK; ++c) {
    size_t base = (size_t)((b * NCHUNK + c) * NHEADS + h);
    size_t idx = base * (HEADDIM * D_STATE) + p * D_STATE + n;
    if (c + 1 < NCHUNK) {  // prefetch next chunk's state line (global_prefetch_b8)
      size_t nxt = (size_t)((b * NCHUNK + c + 1) * NHEADS + h) * (HEADDIM * D_STATE)
                   + p * D_STATE + n;
      __builtin_prefetch(states + nxt, 0, 1);
    }
    ns[idx] = s;
    float Aend = Acs[base * CHUNK + CHUNK - 1];
    s = s * expf(Aend) + states[idx];
  }
}

// ---------------- kernel 7: Y_off + skip + SiLU gate -> f16 -----------------
__global__ __launch_bounds__(128) void yoff_kernel(
    const _Float16* __restrict__ Chp, const _Float16* __restrict__ xh,
    const float* __restrict__ ns, const float* __restrict__ Acs,
    const float* __restrict__ ydiag, const float* __restrict__ proj,
    const float* __restrict__ Dp, _Float16* __restrict__ yh) {
  const int h = blockIdx.x, c = blockIdx.y, b = blockIdx.z;
  const int tid = threadIdx.x, lane = tid & 31, wv = tid >> 5;
  const int t0 = b * SEQ + c * CHUNK;
  __shared__ _Float16 sNS[HEADDIM * D_STATE];
  __shared__ float sAcs[CHUNK];
  const size_t nbase = (size_t)((b * NCHUNK + c) * NHEADS + h) * (HEADDIM * D_STATE);
  for (int idx = tid; idx < HEADDIM * D_STATE; idx += 128)
    sNS[idx] = (_Float16)ns[nbase + idx];
  if (tid < CHUNK)
    sAcs[tid] = Acs[(size_t)((b * NCHUNK + c) * NHEADS + h) * CHUNK + tid];
  __syncthreads();
  // Y_off = C * new_states^T   (64i x 64p, K = 128n)
  const int iT = wv * 16;
  v8f acc[4] = {};
  for (int kt = 0; kt < D_STATE; kt += 32) {
    v16h aC = load_frag(Chp + (size_t)(t0 + iT) * D_STATE + kt, D_STATE, 1);
#pragma unroll
    for (int pt = 0; pt < 4; ++pt) {
      v16h bN = load_frag(sNS + pt * 16 * D_STATE + kt, D_STATE, 1);
      acc[pt] = wmma16(aC, bN, acc[pt]);
    }
  }
  const int rofs = (lane >> 4) * 8, cofs = lane & 15;
  const float Dh = Dp[h];
#pragma unroll
  for (int pt = 0; pt < 4; ++pt)
#pragma unroll
    for (int r = 0; r < 8; ++r) {
      int i = iT + rofs + r, p = pt * 16 + cofs;
      size_t e = (size_t)(t0 + i) * D_INNER + h * HEADDIM + p;
      float yv = ydiag[e] + acc[pt][r] * expf(sAcs[i]) + (float)xh[e] * Dh;
      float z = proj[(size_t)(t0 + i) * D_PROJ + h * HEADDIM + p];  // z slice
      yh[e] = (_Float16)(yv * z / (1.f + expf(-z)));                 // y * silu(z)
    }
}

// ---------------- host launcher ----------------
extern "C" void kernel_launch(void* const* d_in, const int* in_sizes, int n_in,
                              void* d_out, int out_size, void* d_ws, size_t ws_size,
                              hipStream_t stream) {
  (void)in_sizes; (void)n_in; (void)out_size; (void)ws_size;
  const float* u    = (const float*)d_in[0];
  const float* Wp   = (const float*)d_in[1];  // (D_PROJ, DIM)
  const float* Wo   = (const float*)d_in[2];  // (DIM, D_INNER)
  const float* Dp   = (const float*)d_in[3];
  const float* dtb  = (const float*)d_in[4];
  const float* Alog = (const float*)d_in[5];
  const float* bnw  = (const float*)d_in[6];
  const float* cnw  = (const float*)d_in[7];
  const float* lnw  = (const float*)d_in[8];
  const float* lnb  = (const float*)d_in[9];
  float* out = (float*)d_out;

  char* ws = (char*)d_ws;
  size_t cur = 0;
  auto alloc = [&](size_t bytes) -> void* {
    void* p = ws + cur;
    cur += (bytes + 255) & ~(size_t)255;
    return p;
  };
  _Float16* unh  = (_Float16*)alloc((size_t)NTOK * DIM * 2);
  _Float16* Wph  = (_Float16*)alloc((size_t)D_PROJ * DIM * 2);
  _Float16* Woh  = (_Float16*)alloc((size_t)DIM * D_INNER * 2);
  float*    proj = (float*)alloc((size_t)NTOK * D_PROJ * 4);
  _Float16* xh   = (_Float16*)alloc((size_t)NTOK * D_INNER * 2);
  _Float16* Bh   = (_Float16*)alloc((size_t)NTOK * D_STATE * 2);
  _Float16* Ch   = (_Float16*)alloc((size_t)NTOK * D_STATE * 2);
  float*    dA   = (float*)alloc((size_t)NTOK * NHEADS * 4);
  float*    Acs  = (float*)alloc((size_t)BATCH * NCHUNK * NHEADS * CHUNK * 4);
  float*    ydg  = (float*)alloc((size_t)NTOK * D_INNER * 4);
  float*    st   = (float*)alloc((size_t)BATCH * NCHUNK * NHEADS * HEADDIM * D_STATE * 4);
  float*    nst  = (float*)alloc((size_t)BATCH * NCHUNK * NHEADS * HEADDIM * D_STATE * 4);
  _Float16* yh   = (_Float16*)alloc((size_t)NTOK * D_INNER * 2);

  // 1) layernorm -> f16 activations
  layernorm_kernel<<<NTOK, 256, 0, stream>>>(u, lnw, lnb, unh);
  // 2) weight conversions
  {
    int n1 = D_PROJ * DIM;
    cvt_f16_kernel<<<(n1 + 255) / 256, 256, 0, stream>>>(Wp, Wph, n1);
    int n2 = DIM * D_INNER;
    cvt_f16_kernel<<<(n2 + 255) / 256, 256, 0, stream>>>(Wo, Woh, n2);
  }
  // 3) in_proj GEMM:  proj[NTOK, D_PROJ] = u_n * Wp^T
  gemm_f16_kernel<<<dim3(D_PROJ / 64, NTOK / 64), 128, 0, stream>>>(
      unh, Wph, nullptr, proj, NTOK, D_PROJ, DIM);
  // 4) split + rmsnorm + softplus
  extract_kernel<<<NTOK, 128, 0, stream>>>(proj, bnw, cnw, dtb, Alog, xh, Bh, Ch, dA);
  // 5) per-chunk SSD
  chunk_kernel<<<dim3(NHEADS, NCHUNK, BATCH), 128, 0, stream>>>(
      xh, Bh, Ch, dA, Acs, ydg, st);
  // 6) inter-chunk scan
  scan_kernel<<<(BATCH * NHEADS * HEADDIM * D_STATE) / 256, 256, 0, stream>>>(
      st, Acs, nst);
  // 7) Y_off + gate
  yoff_kernel<<<dim3(NHEADS, NCHUNK, BATCH), 128, 0, stream>>>(
      Ch, xh, nst, Acs, ydg, proj, Dp, yh);
  // 8) out_proj GEMM + residual
  gemm_f16_kernel<<<dim3(DIM / 64, NTOK / 64), 128, 0, stream>>>(
      yh, Woh, u, out, NTOK, DIM, D_INNER);
}
